// GateUpv_3_35828617183603
// MI455X (gfx1250) — matrix operand order repaired and checked by reading, not compiled, not run on hardware
//
#include <hip/hip_runtime.h>

// ---------------------------------------------------------------------------
// GateUpv_3 pipeline for MI455X (gfx1250, wave32, WMMA bf16 16x16x32)
//   - all conv layers: implicit GEMM, bf16 A/B, fp32 accum
//   - weights pre-packed to fragment order (A frags load straight from global)
//   - activations between convs stored bf16
//   - one wave computes a 32(co) x 16(pixel) tile: two WMMA acc tiles share
//     one LDS input slab (doubles WMMA per staged byte)
// ---------------------------------------------------------------------------

typedef __attribute__((ext_vector_type(16))) __bf16 v16bf;
typedef __attribute__((ext_vector_type(8)))  __bf16 v8bf;
typedef __attribute__((ext_vector_type(8)))  float  v8f;

#define NB   4
#define HY   65
#define WY   65
#define HW   (HY * WY)      // 4225
#define KD   32
#define EPSB 1e-5f

#define SHUF16(a, b) __builtin_shufflevector(a, b, 0,1,2,3,4,5,6,7,8,9,10,11,12,13,14,15)

// ---------------------------------------------------------------------------
// fp32 -> bf16 elementwise
// ---------------------------------------------------------------------------
__global__ void k_tobf16(const float* __restrict__ in, __bf16* __restrict__ out,
                         long long n)
{
    long long i = (long long)blockIdx.x * blockDim.x + threadIdx.x;
    if (i < n) out[i] = (__bf16)in[i];
}

// ---------------------------------------------------------------------------
// Pack conv weights into WMMA-A fragment order, bf16.
//   dst linear index i = (((cot*nchunks + cic)*9 + t)*512 + m*32 + k)
//   normal:     W[co][ci][kh][kw]   (co = cot*16+m, ci = cic*32+k, t = kh*3+kw)
//   transposed: W[ci][co][kh][kw]   (conv-transpose weights)
// ---------------------------------------------------------------------------
__global__ void k_pack(const float* __restrict__ W, __bf16* __restrict__ dst,
                       int Cin, int Cout, int transposed, int total)
{
    int i = blockIdx.x * 256 + threadIdx.x;
    if (i >= total) return;
    int k    = i & 31;
    int m    = (i >> 5) & 15;
    int t    = (i >> 9) % 9;
    int rest = (i >> 9) / 9;
    int nch  = Cin >> 5;
    int cic  = rest % nch;
    int cot  = rest / nch;
    float s = transposed
            ? W[((size_t)(cic * 32 + k) * Cout + (cot * 16 + m)) * 9 + t]
            : W[((size_t)(cot * 16 + m) * Cin + (cic * 32 + k)) * 9 + t];
    dst[i] = (__bf16)s;
}

// ---------------------------------------------------------------------------
// Generic 3x3 SAME conv, implicit GEMM, one wave per 32(co) x 16(pixel) tile.
// in: bf16 activations.  Wp: packed bf16 weights.  out: fp32.
// mode 0: raw (BN follows).  mode 1: +bias, ReLU.
// ---------------------------------------------------------------------------
__global__ __launch_bounds__(32)
void k_conv3(const __bf16* __restrict__ in, const __bf16* __restrict__ Wp,
             const float* __restrict__ bias, float* __restrict__ out,
             int Cin, int Cout, int mode)
{
    const int lane = threadIdx.x;
    const int ow0  = blockIdx.x * 16;
    const int oh   = blockIdx.y % HY;
    const int b    = blockIdx.y / HY;
    const int cot0 = blockIdx.z * 2;            // two 16-row A tiles per wave
    const int co0  = cot0 * 16;
    const int nch  = Cin >> 5;
    const size_t AtileStride = (size_t)nch * 9 * 512;   // elems between co-tiles

    // slab: [px 0..17][kh 0..2][k 0..31], k fastest (frag = 8 halves at k=8*hi + 8 at k=16+8*hi)
    __shared__ __bf16 Bs[18 * 3 * 32];

    const int m  = lane & 15;
    const int hi = lane >> 4;
    const int kb = hi << 4;

    v8f acc0 = {};
    v8f acc1 = {};

    for (int cic = 0; cic < nch; ++cic) {
        const int ci0 = cic * 32;
        const __bf16* Abase0 = Wp + ((size_t)(cot0 * nch + cic) * 9) * 512;
        const __bf16* Abase1 = Abase0 + AtileStride;
        if (cic + 1 < nch) {
            __builtin_prefetch(Abase0 + 9 * 512, 0, 1);                      // next weights
            __builtin_prefetch(in + ((size_t)(b * Cin + ci0 + 32) * HY + oh) * WY, 0, 1);
        }

        // ---- stage slab: 1728 halves, 54/lane, px-fastest (coalesced) ----
        // element e = lane + 32*i ; dims (px=18, r=3, k=32), px fastest
        __bf16 tmp[54];
        {
            int px = (lane < 18) ? lane : lane - 18;
            int r  = (lane < 18) ? 0 : 1;
            int k  = 0;
            #pragma unroll
            for (int i = 0; i < 54; ++i) {
                int ih = oh + r - 1;
                int iw = ow0 + px - 1;
                __bf16 v = (__bf16)0.0f;
                if (ih >= 0 && ih < HY && iw >= 0 && iw < WY)
                    v = in[((size_t)(b * Cin + ci0 + k) * HY + ih) * WY + iw];
                tmp[i] = v;
                // advance e += 32 over dims (18,3,*)
                px += 14; int c = (px >= 18); if (c) px -= 18;
                r += 1 + c; if (r >= 3) { r -= 3; k += 1; }
            }
        }
        __syncthreads();
        {
            int px = (lane < 18) ? lane : lane - 18;
            int r  = (lane < 18) ? 0 : 1;
            int k  = 0;
            #pragma unroll
            for (int i = 0; i < 54; ++i) {
                Bs[(px * 3 + r) * 32 + k] = tmp[i];
                px += 14; int c = (px >= 18); if (c) px -= 18;
                r += 1 + c; if (r >= 3) { r -= 3; k += 1; }
            }
        }
        __syncthreads();

        // ---- 9 taps, two WMMAs each (co tile 0 and 1 share the B frag) ----
        #pragma unroll
        for (int t = 0; t < 9; ++t) {
            const int kh = t / 3, kw = t % 3;

            const v8bf* ap0 = (const v8bf*)(Abase0 + t * 512 + m * 32);
            v16bf a0 = SHUF16(ap0[hi], ap0[2 + hi]);
            const v8bf* ap1 = (const v8bf*)(Abase1 + t * 512 + m * 32);
            v16bf a1 = SHUF16(ap1[hi], ap1[2 + hi]);

            const v8bf* bp = (const v8bf*)&Bs[((m + kw) * 3 + kh) * 32];
            v16bf bm = SHUF16(bp[hi], bp[2 + hi]);

            acc0 = __builtin_amdgcn_wmma_f32_16x16x32_bf16(
                       false, a0, false, bm, (short)0, acc0, false, false);
            acc1 = __builtin_amdgcn_wmma_f32_16x16x32_bf16(
                       false, a1, false, bm, (short)0, acc1, false, false);
        }
        __syncthreads();
    }

    // epilogue: C/D layout  m_out = 8*hi + r, n = lane&15
    const int n  = lane & 15;
    const int mb = hi * 8;
    const int ow = ow0 + n;
    if (ow < WY) {
        #pragma unroll
        for (int r = 0; r < 8; ++r) {
            int co = co0 + mb + r;
            float v0 = acc0[r];
            float v1 = acc1[r];
            if (mode == 1) {
                v0 += bias[co];      v0 = v0 > 0.f ? v0 : 0.f;
                v1 += bias[co + 16]; v1 = v1 > 0.f ? v1 : 0.f;
            }
            out[((size_t)(b * Cout + co)      * HY + oh) * WY + ow] = v0;
            out[((size_t)(b * Cout + co + 16) * HY + oh) * WY + ow] = v1;
        }
    }
}

// ---------------------------------------------------------------------------
// ConvTranspose2d(256,256,3,stride 2) + bias + ReLU, implicit gather GEMM.
// x: bf16 (pre-converted).  Wp: packed (transposed) weights.  out: bf16.
// One wave per 32(co) x 16(pixel) tile.
// ---------------------------------------------------------------------------
__global__ __launch_bounds__(32)
void k_convt(const __bf16* __restrict__ x, const __bf16* __restrict__ Wp,
             const float* __restrict__ bt, __bf16* __restrict__ out)
{
    const int Cin = 256, Cout = 256, Hin = 32, Win = 32;
    const int lane = threadIdx.x;
    const int ow0  = blockIdx.x * 16;
    const int oh   = blockIdx.y % HY;
    const int b    = blockIdx.y / HY;
    const int cot0 = blockIdx.z * 2;
    const int co0  = cot0 * 16;
    const int nch  = Cin >> 5;
    const size_t AtileStride = (size_t)nch * 9 * 512;

    // slab: [px 0..9][j 0..1][k 0..31], k fastest.  j = input-row slot.
    __shared__ __bf16 Bs[10 * 2 * 32];

    const int m  = lane & 15;
    const int hi = lane >> 4;
    const int kb = hi << 4;
    const int ohEven = ((oh & 1) == 0);
    const int cbase  = (ow0 >> 1) - 1;           // first staged input column

    v8f acc0 = {};
    v8f acc1 = {};

    for (int cic = 0; cic < nch; ++cic) {
        const int ci0 = cic * 32;
        const __bf16* Abase0 = Wp + ((size_t)(cot0 * nch + cic) * 9) * 512;
        const __bf16* Abase1 = Abase0 + AtileStride;
        if (cic + 1 < nch) __builtin_prefetch(Abase0 + 9 * 512, 0, 1);

        // ---- stage slab: 640 halves, 20/lane, px-fastest ----
        // element e = lane + 32*i ; dims (px=10, j=2, k=32)
        __bf16 tmp[20];
        {
            int px = lane % 10;
            int t2 = lane / 10;                  // t2 = j + 2*k
            #pragma unroll
            for (int i = 0; i < 20; ++i) {
                int j   = t2 & 1;
                int k   = t2 >> 1;
                int col = cbase + px;
                int row = ohEven ? ((oh >> 1) - j) : ((oh - 1) >> 1);
                bool ok = (ohEven || j == 0) && row >= 0 && row < Hin
                          && col >= 0 && col < Win;
                __bf16 v = (__bf16)0.0f;
                if (ok)
                    v = x[((size_t)(b * Cin + ci0 + k) * Hin + row) * Win + col];
                tmp[i] = v;
                px += 2; int c = (px >= 10); if (c) px -= 10;
                t2 += 3 + c;
            }
        }
        __syncthreads();
        {
            int px = lane % 10;
            int t2 = lane / 10;
            #pragma unroll
            for (int i = 0; i < 20; ++i) {
                int j = t2 & 1;
                int k = t2 >> 1;
                Bs[(px * 2 + j) * 32 + k] = tmp[i];
                px += 2; int c = (px >= 10); if (c) px -= 10;
                t2 += 3 + c;
            }
        }
        __syncthreads();

        // ---- taps: only kh matching oh parity contribute (uniform skip) ----
        #pragma unroll
        for (int kh = 0; kh < 3; ++kh) {
            int j;
            if (ohEven) { if (kh == 1) continue; j = (kh == 2); }
            else        { if (kh != 1) continue; j = 0; }
            #pragma unroll
            for (int kw = 0; kw < 3; ++kw) {
                const int t = kh * 3 + kw;
                const v8bf* ap0 = (const v8bf*)(Abase0 + t * 512 + m * 32);
                v16bf a0 = SHUF16(ap0[hi], ap0[2 + hi]);
                const v8bf* ap1 = (const v8bf*)(Abase1 + t * 512 + m * 32);
                v16bf a1 = SHUF16(ap1[hi], ap1[2 + hi]);

                v16bf bm = {};
                int iww = ow0 + m - kw;          // m == n for B columns
                if (iww >= 0 && ((iww & 1) == 0) && ((iww >> 1) < Win)) {
                    int px = (iww >> 1) - cbase;             // 0..8
                    const v8bf* bp = (const v8bf*)&Bs[(px * 2 + j) * 32];
                    bm = SHUF16(bp[hi], bp[2 + hi]);
                }
                acc0 = __builtin_amdgcn_wmma_f32_16x16x32_bf16(
                           false, a0, false, bm, (short)0, acc0, false, false);
                acc1 = __builtin_amdgcn_wmma_f32_16x16x32_bf16(
                           false, a1, false, bm, (short)0, acc1, false, false);
            }
        }
        __syncthreads();
    }

    const int n  = lane & 15;
    const int mb = hi * 8;
    const int ow = ow0 + n;
    if (ow < WY) {
        #pragma unroll
        for (int r = 0; r < 8; ++r) {
            int co = co0 + mb + r;
            float v0 = acc0[r] + bt[co];
            float v1 = acc1[r] + bt[co + 16];
            out[((size_t)(b * Cout + co)      * HY + oh) * WY + ow] =
                (__bf16)(v0 > 0.f ? v0 : 0.f);
            out[((size_t)(b * Cout + co + 16) * HY + oh) * WY + ow] =
                (__bf16)(v1 > 0.f ? v1 : 0.f);
        }
    }
}

// ---------------------------------------------------------------------------
// Per-channel batch statistics -> scale/shift (training-mode BN, ddof=0)
// ---------------------------------------------------------------------------
__global__ __launch_bounds__(256)
void k_stats(const float* __restrict__ in, const float* __restrict__ gam,
             const float* __restrict__ bet, float* __restrict__ scale,
             float* __restrict__ shift, int C)
{
    const int c   = blockIdx.x;
    const int tid = threadIdx.x;
    float s = 0.f, s2 = 0.f;
    for (int b = 0; b < NB; ++b) {
        const float* p = in + (size_t)(b * C + c) * HW;
        for (int i = tid; i < HW; i += 256) {
            float v = p[i];
            s += v; s2 += v * v;
        }
    }
    __shared__ float sh[256], sh2[256];
    sh[tid] = s; sh2[tid] = s2;
    __syncthreads();
    for (int o = 128; o > 0; o >>= 1) {
        if (tid < o) { sh[tid] += sh[tid + o]; sh2[tid] += sh2[tid + o]; }
        __syncthreads();
    }
    if (tid == 0) {
        const float inv_n = 1.0f / (float)(NB * HW);
        float mu  = sh[0] * inv_n;
        float var = sh2[0] * inv_n - mu * mu;
        float sc  = gam[c] * rsqrtf(var + EPSB);
        scale[c] = sc;
        shift[c] = bet[c] - mu * sc;
    }
}

// ---------------------------------------------------------------------------
// BN apply + ReLU.  out_f32 ? fp32 : bf16 output.
// ---------------------------------------------------------------------------
__global__ void k_bnrelu(const float* __restrict__ in, const float* __restrict__ scale,
                         const float* __restrict__ shift, void* __restrict__ out,
                         int C, long long total, int out_f32)
{
    long long i = (long long)blockIdx.x * blockDim.x + threadIdx.x;
    if (i >= total) return;
    int c = (int)((i / HW) % C);
    float v = in[i] * scale[c] + shift[c];
    v = v > 0.f ? v : 0.f;
    if (out_f32) ((float*)out)[i] = v;
    else         ((__bf16*)out)[i] = (__bf16)v;
}

// ---------------------------------------------------------------------------
// Depthwise 3x3 SAME + bias + ReLU  (Q path), fp32
// ---------------------------------------------------------------------------
__global__ void k_qdw(const float* __restrict__ y, const float* __restrict__ Wq,
                      const float* __restrict__ bq, float* __restrict__ q,
                      long long total)
{
    long long i = (long long)blockIdx.x * blockDim.x + threadIdx.x;
    if (i >= total) return;
    int w = (int)(i % WY);
    int h = (int)((i / WY) % HY);
    int c = (int)((i / HW) % 128);
    int b = (int)(i / ((long long)HW * 128));
    const float* yp = y + (size_t)(b * 128 + c) * HW;
    float s = 0.f;
    #pragma unroll
    for (int kh = 0; kh < 3; ++kh) {
        int ih = h + kh - 1;
        if (ih < 0 || ih >= HY) continue;
        #pragma unroll
        for (int kw = 0; kw < 3; ++kw) {
            int iw = w + kw - 1;
            if (iw < 0 || iw >= WY) continue;
            s += yp[ih * WY + iw] * Wq[c * 9 + kh * 3 + kw];
        }
    }
    s += bq[c];
    q[i] = s > 0.f ? s : 0.f;
}

// ---------------------------------------------------------------------------
// Gate: softmax over KD keys per (b,c,h,w); V = relu(y*Wv+bv).  Output bf16.
// ---------------------------------------------------------------------------
__global__ void k_gate(const float* __restrict__ q, const float* __restrict__ kk,
                       const float* __restrict__ y, const float* __restrict__ Wv,
                       const float* __restrict__ bv, __bf16* __restrict__ g,
                       long long total)
{
    long long i = (long long)blockIdx.x * blockDim.x + threadIdx.x;
    if (i >= total) return;
    int sp = (int)(i % HW);
    int b  = (int)(i / ((long long)HW * 128));
    float qv = q[i];
    float yv = y[i];
    const float* kp = kk + (size_t)b * KD * HW + sp;

    float s[KD];
    float mx = -3.4e38f;
    #pragma unroll
    for (int d = 0; d < KD; ++d) {
        s[d] = qv * kp[(size_t)d * HW];
        mx = s[d] > mx ? s[d] : mx;
    }
    float se = 0.f, acc = 0.f;
    #pragma unroll
    for (int d = 0; d < KD; ++d) {
        float e  = __expf(s[d] - mx);
        float vv = yv * Wv[d] + bv[d];
        vv = vv > 0.f ? vv : 0.f;
        se  += e;
        acc += e * vv;
    }
    g[i] = (__bf16)(acc / se);
}

// ---------------------------------------------------------------------------
// launch
// ---------------------------------------------------------------------------
extern "C" void kernel_launch(void* const* d_in, const int* in_sizes, int n_in,
                              void* d_out, int out_size, void* d_ws, size_t ws_size,
                              hipStream_t stream)
{
    const float* x  = (const float*)d_in[0];
    const float* y  = (const float*)d_in[1];
    const float* Wt = (const float*)d_in[2];
    const float* bt = (const float*)d_in[3];
    const float* W1 = (const float*)d_in[4];
    const float* g1 = (const float*)d_in[5];
    const float* b1 = (const float*)d_in[6];
    const float* W2 = (const float*)d_in[7];
    const float* g2 = (const float*)d_in[8];
    const float* b2 = (const float*)d_in[9];
    const float* Wq = (const float*)d_in[10];
    const float* bq = (const float*)d_in[11];
    const float* Wk = (const float*)d_in[12];
    const float* bk = (const float*)d_in[13];
    const float* Wv = (const float*)d_in[14];
    const float* bv = (const float*)d_in[15];
    const float* W3 = (const float*)d_in[16];
    const float* g3 = (const float*)d_in[17];
    const float* b3 = (const float*)d_in[18];
    const float* W4 = (const float*)d_in[19];
    const float* g4 = (const float*)d_in[20];
    const float* b4 = (const float*)d_in[21];

    // ---- workspace carve (256B aligned regions) ----
    char* cur = (char*)d_ws;
    auto carve = [&](size_t bytes) {
        char* r = cur;
        cur += (bytes + 255) & ~(size_t)255;
        return (void*)r;
    };
    const size_t NX  = (size_t)NB * 256 * 32 * 32;   // 1,048,576
    const size_t NH  = (size_t)NB * 256 * HW;        // 4,326,400
    const size_t NC  = (size_t)NB * 128 * HW;        // 2,163,200
    const size_t NKk = (size_t)NB * KD  * HW;        //   540,800

    __bf16* XBF = (__bf16*)carve(NX * 2);
    __bf16* WtP = (__bf16*)carve((size_t)256 * 256 * 9 * 2);
    __bf16* W1P = (__bf16*)carve((size_t)128 * 256 * 9 * 2);
    __bf16* W2P = (__bf16*)carve((size_t)128 * 128 * 9 * 2);
    __bf16* WkP = (__bf16*)carve((size_t)32  * 128 * 9 * 2);
    __bf16* W3P = (__bf16*)carve((size_t)128 * 128 * 9 * 2);
    __bf16* W4P = (__bf16*)carve((size_t)128 * 128 * 9 * 2);
    __bf16* Hbf = (__bf16*)carve(NH * 2);            // h; halves reused as U1/U2
    float*  R   = (float*)carve(NC * 4);             // raw conv out (fp32)
    float*  Q   = (float*)carve(NC * 4);
    float*  KK  = (float*)carve(NKk * 4);
    float*  SC  = (float*)carve(128 * 4);
    float*  SH  = (float*)carve(128 * 4);
    __bf16* U1  = Hbf;                               // alias lower half
    __bf16* U2  = Hbf + NC;                          // alias upper half

    const dim3 wv(32);
    const long long tot = (long long)NC;
    const unsigned nbE  = (unsigned)((tot + 255) / 256);

    // ---- prep: bf16 x, packed weights ----
    k_tobf16<<<(unsigned)((NX + 255) / 256), 256, 0, stream>>>(x, XBF, (long long)NX);
    k_pack<<<(256*256*9 + 255) / 256, 256, 0, stream>>>(Wt, WtP, 256, 256, 1, 256*256*9);
    k_pack<<<(128*256*9 + 255) / 256, 256, 0, stream>>>(W1, W1P, 256, 128, 0, 128*256*9);
    k_pack<<<(128*128*9 + 255) / 256, 256, 0, stream>>>(W2, W2P, 128, 128, 0, 128*128*9);
    k_pack<<<(32 *128*9 + 255) / 256, 256, 0, stream>>>(Wk, WkP, 128, 32,  0, 32*128*9);
    k_pack<<<(128*128*9 + 255) / 256, 256, 0, stream>>>(W3, W3P, 128, 128, 0, 128*128*9);
    k_pack<<<(128*128*9 + 255) / 256, 256, 0, stream>>>(W4, W4P, 128, 128, 0, 128*128*9);

    // 1. h = relu(convT(x) + bt)                                   -> Hbf (bf16)
    k_convt<<<dim3(5, NB * HY, 8), wv, 0, stream>>>(XBF, WtP, bt, Hbf);
    // 2. conv1 raw 256->128                                         -> R
    k_conv3<<<dim3(5, NB * HY, 4), wv, 0, stream>>>(Hbf, W1P, nullptr, R, 256, 128, 0);
    k_stats<<<128, 256, 0, stream>>>(R, g1, b1, SC, SH, 128);
    k_bnrelu<<<nbE, 256, 0, stream>>>(R, SC, SH, U1, 128, tot, 0);
    // 3. conv2 raw 128->128                                         -> R
    k_conv3<<<dim3(5, NB * HY, 4), wv, 0, stream>>>(U1, W2P, nullptr, R, 128, 128, 0);
    k_stats<<<128, 256, 0, stream>>>(R, g2, b2, SC, SH, 128);
    k_bnrelu<<<nbE, 256, 0, stream>>>(R, SC, SH, U2, 128, tot, 0);     // xu -> U2
    // 4. K = relu(conv(xu, Wk) + bk)  128->32                       -> KK (fp32)
    k_conv3<<<dim3(5, NB * HY, 1), wv, 0, stream>>>(U2, WkP, bk, KK, 128, 32, 1);
    // 5. q = relu(dwconv(y, Wq) + bq)                               -> Q (fp32)
    k_qdw<<<nbE, 256, 0, stream>>>(y, Wq, bq, Q, tot);
    // 6. g = softmax-gate(q, K, y)                                  -> U1 (bf16)
    k_gate<<<nbE, 256, 0, stream>>>(Q, KK, y, Wv, bv, U1, tot);
    // 7. conv3 raw 128->128                                         -> R
    k_conv3<<<dim3(5, NB * HY, 4), wv, 0, stream>>>(U1, W3P, nullptr, R, 128, 128, 0);
    k_stats<<<128, 256, 0, stream>>>(R, g3, b3, SC, SH, 128);
    k_bnrelu<<<nbE, 256, 0, stream>>>(R, SC, SH, U2, 128, tot, 0);
    // 8. conv4 raw 128->128                                         -> R
    k_conv3<<<dim3(5, NB * HY, 4), wv, 0, stream>>>(U2, W4P, nullptr, R, 128, 128, 0);
    k_stats<<<128, 256, 0, stream>>>(R, g4, b4, SC, SH, 128);
    k_bnrelu<<<nbE, 256, 0, stream>>>(R, SC, SH, d_out, 128, tot, 1);
}
